// UltraEnhancedHRM_74311524155950
// MI455X (gfx1250) — compile-verified, matrix operand order checked by reading, not run on hardware
//
#include <hip/hip_runtime.h>
#include <hip/hip_bf16.h>

typedef __attribute__((ext_vector_type(16))) _Float16 v16h;
typedef __attribute__((ext_vector_type(8)))  _Float16 v8h;
typedef __attribute__((ext_vector_type(8)))  float    v8f;

#define HDIM   256
#define LSEQ   256
#define BATCH  32
#define NHEAD  8
#define DHEAD  32
#define NLAYER 4
#define NTOK   (BATCH * LSEQ)   // 8192

__device__ __forceinline__ float gelu_exact(float x) {
    return 0.5f * x * (1.0f + erff(x * 0.70710678118654752f));
}

// ---------------------------------------------------------------------------
// Tiled weight convert+transpose: W[K,N] f32 -> Wt[N,K] f16.
// 32x32 tiles through LDS; coalesced on both sides (K,N multiples of 32).
// ---------------------------------------------------------------------------
__global__ void k_transpose_tiled(const float* __restrict__ W, _Float16* __restrict__ Wt,
                                  int K, int N) {
    __shared__ _Float16 tile[32][33];
    int k0 = blockIdx.x * 32;
    int n0 = blockIdx.y * 32;
    int c = threadIdx.x & 31;
    int r8 = threadIdx.x >> 5;      // 0..7
    #pragma unroll
    for (int t = 0; t < 4; ++t) {
        int r = r8 + t * 8;
        tile[r][c] = (_Float16)W[(long long)(k0 + r) * N + n0 + c];
    }
    __syncthreads();
    #pragma unroll
    for (int t = 0; t < 4; ++t) {
        int r = r8 + t * 8;
        Wt[(long long)(n0 + r) * K + k0 + c] = tile[c][r];
    }
}

// ---------------------------------------------------------------------------
// V transpose for attention: kv [NTOK,512] (V = cols 256..511)
//   -> vt [(b*8+h)*32 + d][j]  (f16, per-(b,h) 32x256 row-major over keys)
// ---------------------------------------------------------------------------
__global__ void k_transpose_v(const _Float16* __restrict__ kv, _Float16* __restrict__ vt) {
    int idx = blockIdx.x * 256 + threadIdx.x;   // 0 .. NTOK*256-1
    int j = idx & 255;
    int rest = idx >> 8;        // (b*8+h)*32 + d
    int d = rest & 31;
    int bh = rest >> 5;
    int b = bh >> 3, h = bh & 7;
    vt[(long long)rest * LSEQ + j] =
        kv[((long long)(b * LSEQ + j)) * (2 * HDIM) + HDIM + h * DHEAD + d];
}

// ---------------------------------------------------------------------------
// Embedding: build f16 concat input [NTOK, 544]
// ---------------------------------------------------------------------------
__global__ void k_embed(const int* __restrict__ loc, const int* __restrict__ start_min,
                        const int* __restrict__ weekday, const int* __restrict__ user,
                        const float* __restrict__ duration,
                        const float* __restrict__ fine_E, const float* __restrict__ coarse_E,
                        const float* __restrict__ hour_E, const float* __restrict__ wd_E,
                        const float* __restrict__ Wt, const float* __restrict__ bt,
                        const float* __restrict__ lnt_g, const float* __restrict__ lnt_b,
                        const float* __restrict__ user_E, const float* __restrict__ Wd,
                        const float* __restrict__ bd,
                        _Float16* __restrict__ cin) {
    int t = blockIdx.x;      // token
    int h = threadIdx.x;     // 0..255
    int b = t / LSEQ;
    int lid = loc[t];

    int sm = start_min[t];
    int hour = (sm / 60) % 24;
    float minute = (float)(sm % 60) * (1.0f / 60.0f);
    int wd = weekday[t];

    float acc = bt[h];
    #pragma unroll
    for (int j = 0; j < 12; ++j) acc += hour_E[hour * 12 + j] * Wt[j * HDIM + h];
    #pragma unroll
    for (int j = 0; j < 8; ++j)  acc += wd_E[wd * 8 + j] * Wt[(12 + j) * HDIM + h];
    acc += minute * Wt[20 * HDIM + h];
    float gv = gelu_exact(acc);

    __shared__ float red[256];
    red[h] = gv; __syncthreads();
    for (int s = 128; s > 0; s >>= 1) { if (h < s) red[h] += red[h + s]; __syncthreads(); }
    float mean = red[0] * (1.0f / 256.0f);
    __syncthreads();
    float dc = gv - mean;
    red[h] = dc * dc; __syncthreads();
    for (int s = 128; s > 0; s >>= 1) { if (h < s) red[h] += red[h + s]; __syncthreads(); }
    float var = red[0] * (1.0f / 256.0f);
    float tn = dc * rsqrtf(var + 1e-5f) * lnt_g[h] + lnt_b[h];

    _Float16* row = cin + (long long)t * 544;
    if (h < 128) row[h] = (_Float16)fine_E[(long long)lid * 128 + h];
    else         row[h] = (_Float16)coarse_E[(lid % 64) * 128 + (h - 128)];
    row[256 + h] = (_Float16)tn;
    if (h < 16) {
        row[512 + h] = (_Float16)user_E[user[b * LSEQ] * 16 + h];
    } else if (h < 32) {
        int j = h - 16;
        row[528 + j] = (_Float16)(duration[t] * Wd[j] + bd[j]);
    }
}

// ---------------------------------------------------------------------------
// Row LayerNorm (H=256) f32 -> f16.  final_sel: pick row r*L + (L-1).
// ---------------------------------------------------------------------------
__global__ void k_ln(const float* __restrict__ x, const float* __restrict__ g,
                     const float* __restrict__ bta, _Float16* __restrict__ out,
                     int final_sel) {
    int r = blockIdx.x;
    int h = threadIdx.x;
    long long src = final_sel ? ((long long)r * LSEQ + (LSEQ - 1)) : (long long)r;
    float v = x[src * HDIM + h];
    __shared__ float red[256];
    red[h] = v; __syncthreads();
    for (int s = 128; s > 0; s >>= 1) { if (h < s) red[h] += red[h + s]; __syncthreads(); }
    float mean = red[0] * (1.0f / 256.0f);
    __syncthreads();
    float dc = v - mean;
    red[h] = dc * dc; __syncthreads();
    for (int s = 128; s > 0; s >>= 1) { if (h < s) red[h] += red[h + s]; __syncthreads(); }
    float var = red[0] * (1.0f / 256.0f);
    out[(long long)r * HDIM + h] = (_Float16)(dc * rsqrtf(var + 1e-5f) * g[h] + bta[h]);
}

// ---------------------------------------------------------------------------
// WMMA GEMM with async global->LDS staging of the shared A tile.
//   C[M,N] = epilogue(A[M,K] @ Wt[N,K]^T); A,Wt f16 row-major.
//   flags: 1 = +bias[n], 2 = GELU, 4 = +R[m*N+n]
// ---------------------------------------------------------------------------
__global__ void k_wmma_gemm(const _Float16* __restrict__ A, const _Float16* __restrict__ Wt,
                            const float* __restrict__ bias, const float* __restrict__ R,
                            float* __restrict__ C, _Float16* __restrict__ Ch,
                            int M, int N, int K, int flags) {
    __shared__ __align__(16) _Float16 lA[2][16 * 32];   // 2 x 1KB
    int lane = threadIdx.x & 31;
    int wave = threadIdx.x >> 5;
    int m0 = blockIdx.x * 16;
    int n0 = (blockIdx.y * 8 + wave) * 16;
    if (n0 + 16 > N) n0 = N - 16;   // clamp; keeps barrier participation uniform

    int lane15 = lane & 15;
    int hi = lane >> 4;
    int koff = hi ? 8 : 0;

    const _Float16* Bp = Wt + (long long)(n0 + lane15) * K + koff;

    // wave 0 stages A[m0..m0+15][k0..k0+31] (1KB) into lA[buf]: 2 x b128 per lane
    auto stage = [&](int buf, int k0) {
        if (wave == 0) {
            #pragma unroll
            for (int t = 0; t < 2; ++t) {
                int chunk = lane + 32 * t;      // 0..63 16B chunks
                int row = chunk >> 2;
                int sub = chunk & 3;
                unsigned loff = (unsigned)(size_t)&lA[buf][row * 32 + sub * 8];
                const _Float16* g = A + (long long)(m0 + row) * K + k0 + sub * 8;
                asm volatile("global_load_async_to_lds_b128 %0, %1, off"
                             :: "v"(loff), "v"((unsigned long long)(size_t)g)
                             : "memory");
            }
        }
    };

    stage(0, 0);
    v8f c = {};
    int nk = K >> 5;
    for (int kc = 0; kc < nk; ++kc) {
        int cur = kc & 1;
        int k0 = kc << 5;
        if (kc + 1 < nk) {
            stage(cur ^ 1, k0 + 32);
            asm volatile("s_wait_asynccnt 0x2" ::: "memory");  // current tile landed
        } else {
            asm volatile("s_wait_asynccnt 0x0" ::: "memory");
        }
        __syncthreads();
        union { v16h v; v8h h2[2]; } af, bf;
        af.h2[0] = *(const v8h*)&lA[cur][lane15 * 32 + koff];
        af.h2[1] = *(const v8h*)&lA[cur][lane15 * 32 + koff + 16];
        bf.h2[0] = *(const v8h*)(Bp + k0);
        bf.h2[1] = *(const v8h*)(Bp + k0 + 16);
        if (k0 + 64 < K) __builtin_prefetch(Bp + k0 + 64, 0, 3);
        c = __builtin_amdgcn_wmma_f32_16x16x32_f16(false, af.v, false, bf.v,
                                                   (short)0, c, false, false);
        __syncthreads();    // before cur buffer is overwritten (next iter's stage)
    }

    int n = n0 + lane15;
    float bv = (flags & 1) ? bias[n] : 0.0f;
    int mbase = m0 + 8 * hi;
    #pragma unroll
    for (int r = 0; r < 8; ++r) {
        int m = mbase + r;
        float val = c[r] + bv;
        if (flags & 2) val = gelu_exact(val);
        if (flags & 4) val += R[(long long)m * N + n];
        if (C)  C[(long long)m * N + n]  = val;
        if (Ch) Ch[(long long)m * N + n] = (_Float16)val;
    }
}

// ---------------------------------------------------------------------------
// Output head GEMM: M fixed = 32. Each wave computes BOTH 16-row M-tiles per
// B-fragment load -> WoutT streamed exactly once (HBM-bound floor).
//   C[32,N] = A[32,K] @ Wt[N,K]^T + bias
// ---------------------------------------------------------------------------
__global__ void k_head_gemm(const _Float16* __restrict__ A, const _Float16* __restrict__ Wt,
                            const float* __restrict__ bias, float* __restrict__ C,
                            int N, int K) {
    int lane = threadIdx.x & 31;
    int wave = threadIdx.x >> 5;
    int n0 = (blockIdx.x * 8 + wave) * 16;
    if (n0 + 16 > N) n0 = N - 16;
    int lane15 = lane & 15;
    int hi = lane >> 4;
    int koff = hi ? 8 : 0;

    const _Float16* Ap0 = A + (long long)lane15 * K + koff;          // M-tile 0
    const _Float16* Ap1 = A + (long long)(16 + lane15) * K + koff;   // M-tile 1
    const _Float16* Bp  = Wt + (long long)(n0 + lane15) * K + koff;

    v8f c0 = {}, c1 = {};
    for (int k0 = 0; k0 < K; k0 += 32) {
        union { v16h v; v8h h2[2]; } a0, a1, bf;
        a0.h2[0] = *(const v8h*)(Ap0 + k0);  a0.h2[1] = *(const v8h*)(Ap0 + k0 + 16);
        a1.h2[0] = *(const v8h*)(Ap1 + k0);  a1.h2[1] = *(const v8h*)(Ap1 + k0 + 16);
        bf.h2[0] = *(const v8h*)(Bp + k0);   bf.h2[1] = *(const v8h*)(Bp + k0 + 16);
        if (k0 + 64 < K) __builtin_prefetch(Bp + k0 + 64, 0, 3);
        c0 = __builtin_amdgcn_wmma_f32_16x16x32_f16(false, a0.v, false, bf.v,
                                                    (short)0, c0, false, false);
        c1 = __builtin_amdgcn_wmma_f32_16x16x32_f16(false, a1.v, false, bf.v,
                                                    (short)0, c1, false, false);
    }
    int n = n0 + lane15;
    float bv = bias[n];
    #pragma unroll
    for (int r = 0; r < 8; ++r) {
        C[(long long)(r + 8 * hi) * N + n]      = c0[r] + bv;
        C[(long long)(16 + r + 8 * hi) * N + n] = c1[r] + bv;
    }
}

// ---------------------------------------------------------------------------
// WMMA flash attention, dual path (decay-biased short + plain long), causal.
// ---------------------------------------------------------------------------
__global__ void k_attn_wmma(const _Float16* __restrict__ q,
                            const _Float16* __restrict__ skv,
                            const _Float16* __restrict__ lkv,
                            const _Float16* __restrict__ vst,
                            const _Float16* __restrict__ vlt,
                            const float* __restrict__ mix,
                            const float* __restrict__ decay,
                            int layer, _Float16* __restrict__ ao) {
    __shared__ __align__(16) _Float16 plds[8][16 * 32];  // per-wave P tile (1KB)
    int bh = blockIdx.x;
    int b = bh >> 3, h = bh & 7;
    int lane = threadIdx.x & 31;
    int wave = threadIdx.x >> 5;
    int i0 = blockIdx.y * 128 + wave * 16;
    _Float16* pw = plds[wave];

    int lane15 = lane & 15;
    int hi = lane >> 4;
    int koff = hi ? 8 : 0;

    float alpha = 1.0f / (1.0f + __expf(-mix[layer]));
    float slope = -(1.0f - decay[layer]);
    const float scale = 0.17677669529663687f;   // 1/sqrt(32)

    union V16 { v16h v; v8h h2[2]; };
    const _Float16* qrow = q + (long long)(b * LSEQ + i0 + lane15) * HDIM + h * DHEAD + koff;
    V16 qa;
    qa.h2[0] = *(const v8h*)qrow;
    qa.h2[1] = *(const v8h*)(qrow + 16);

    int nchunk = (i0 + 16 + 31) >> 5;

    v8f oA0 = {}, oA1 = {}, oB0 = {}, oB1 = {};
    float lA[8], lB[8];

    auto run_path = [&](const _Float16* kv, const _Float16* vt, bool biased,
                        v8f& o0, v8f& o1, float* lsum) {
        float m[8];
        #pragma unroll
        for (int r = 0; r < 8; ++r) { m[r] = -3.0e38f; lsum[r] = 0.0f; }
        const _Float16* kb = kv + h * DHEAD + koff;
        const _Float16* vb0 = vt + (long long)(bh * DHEAD + lane15) * LSEQ + koff;
        const _Float16* vb1 = vt + (long long)(bh * DHEAD + 16 + lane15) * LSEQ + koff;
        for (int cch = 0; cch < nchunk; ++cch) {
            int jt = cch << 5;
            const _Float16* kp0 = kb + (long long)(b * LSEQ + jt + lane15) * (2 * HDIM);
            const _Float16* kp1 = kp0 + 16 * (2 * HDIM);
            V16 kf0, kf1;
            kf0.h2[0] = *(const v8h*)kp0;       kf0.h2[1] = *(const v8h*)(kp0 + 16);
            kf1.h2[0] = *(const v8h*)kp1;       kf1.h2[1] = *(const v8h*)(kp1 + 16);
            v8f z = {};
            v8f s0 = __builtin_amdgcn_wmma_f32_16x16x32_f16(false, qa.v, false, kf0.v,
                                                            (short)0, z, false, false);
            v8f s1 = __builtin_amdgcn_wmma_f32_16x16x32_f16(false, qa.v, false, kf1.v,
                                                            (short)0, z, false, false);
            #pragma unroll
            for (int r = 0; r < 8; ++r) {
                int i = i0 + r + 8 * hi;
                int j0 = jt + lane15, j1 = j0 + 16;
                float v0 = s0[r] * scale;
                float v1 = s1[r] * scale;
                if (biased) { v0 += slope * (float)(i - j0); v1 += slope * (float)(i - j1); }
                v0 = (j0 <= i) ? v0 : -3.0e38f;
                v1 = (j1 <= i) ? v1 : -3.0e38f;
                float mx = fmaxf(v0, v1);
                mx = fmaxf(mx, __shfl_xor(mx, 1, 32));
                mx = fmaxf(mx, __shfl_xor(mx, 2, 32));
                mx = fmaxf(mx, __shfl_xor(mx, 4, 32));
                mx = fmaxf(mx, __shfl_xor(mx, 8, 32));
                float mnew = fmaxf(m[r], mx);
                float sc = __expf(m[r] - mnew);
                m[r] = mnew;
                float p0 = __expf(v0 - mnew);   // masked -> exp(-huge) = 0
                float p1 = __expf(v1 - mnew);
                float ps = p0 + p1;
                ps += __shfl_xor(ps, 1, 32);
                ps += __shfl_xor(ps, 2, 32);
                ps += __shfl_xor(ps, 4, 32);
                ps += __shfl_xor(ps, 8, 32);
                lsum[r] = lsum[r] * sc + ps;
                o0[r] *= sc;
                o1[r] *= sc;
                int rowl = r + 8 * hi;
                pw[rowl * 32 + lane15]      = (_Float16)p0;
                pw[rowl * 32 + 16 + lane15] = (_Float16)p1;
            }
            asm volatile("s_wait_dscnt 0x0" ::: "memory");   // wave-local LDS RAW
            V16 pf, vf0, vf1;
            pf.h2[0] = *(const v8h*)(pw + lane15 * 32 + koff);
            pf.h2[1] = *(const v8h*)(pw + lane15 * 32 + koff + 16);
            vf0.h2[0] = *(const v8h*)(vb0 + jt);  vf0.h2[1] = *(const v8h*)(vb0 + jt + 16);
            vf1.h2[0] = *(const v8h*)(vb1 + jt);  vf1.h2[1] = *(const v8h*)(vb1 + jt + 16);
            o0 = __builtin_amdgcn_wmma_f32_16x16x32_f16(false, pf.v, false, vf0.v,
                                                        (short)0, o0, false, false);
            o1 = __builtin_amdgcn_wmma_f32_16x16x32_f16(false, pf.v, false, vf1.v,
                                                        (short)0, o1, false, false);
        }
    };
    run_path(skv, vst, true,  oA0, oA1, lA);
    run_path(lkv, vlt, false, oB0, oB1, lB);

    #pragma unroll
    for (int r = 0; r < 8; ++r) {
        int i = i0 + r + 8 * hi;
        float wA = alpha / lA[r];
        float wB = (1.0f - alpha) / lB[r];
        long long base = (long long)(b * LSEQ + i) * HDIM + h * DHEAD;
        ao[base + lane15]      = (_Float16)(oA0[r] * wA + oB0[r] * wB);
        ao[base + 16 + lane15] = (_Float16)(oA1[r] * wA + oB1[r] * wB);
    }
}

// ---------------------------------------------------------------------------
extern "C" void kernel_launch(void* const* d_in, const int* in_sizes, int n_in,
                              void* d_out, int out_size, void* d_ws, size_t ws_size,
                              hipStream_t stream) {
    (void)in_sizes; (void)n_in; (void)out_size; (void)ws_size;
    const int*   loc      = (const int*)  d_in[0];
    const int*   start_mn = (const int*)  d_in[1];
    const int*   weekday  = (const int*)  d_in[2];
    const int*   user     = (const int*)  d_in[3];
    const float* duration = (const float*)d_in[4];
    const float* fine_E   = (const float*)d_in[5];
    const float* coarse_E = (const float*)d_in[6];
    const float* hour_E   = (const float*)d_in[7];
    const float* wd_E     = (const float*)d_in[8];
    const float* Wtc      = (const float*)d_in[9];
    const float* bt       = (const float*)d_in[10];
    const float* lnt_g    = (const float*)d_in[11];
    const float* lnt_b    = (const float*)d_in[12];
    const float* user_E   = (const float*)d_in[13];
    const float* Wd       = (const float*)d_in[14];
    const float* bd       = (const float*)d_in[15];
    const float* Win      = (const float*)d_in[16];
    const float* bin      = (const float*)d_in[17];
    const float* Wq       = (const float*)d_in[18];
    const float* Wskv     = (const float*)d_in[19];
    const float* Wlkv     = (const float*)d_in[20];
    const float* mix      = (const float*)d_in[21];
    const float* decay    = (const float*)d_in[22];
    const float* Wo       = (const float*)d_in[23];
    const float* ln_g     = (const float*)d_in[24];
    const float* ln_b     = (const float*)d_in[25];
    const float* W1       = (const float*)d_in[26];
    const float* b1       = (const float*)d_in[27];
    const float* W2       = (const float*)d_in[28];
    const float* b2       = (const float*)d_in[29];
    const float* out_g    = (const float*)d_in[30];
    const float* out_b    = (const float*)d_in[31];
    const float* Wout     = (const float*)d_in[32];
    const float* bout     = (const float*)d_in[33];
    float* logits = (float*)d_out;

    const int V = 100000;
    char* ws = (char*)d_ws;
    size_t off = 0;
    auto carve = [&](size_t bytes) -> void* {
        void* p = (void*)(ws + off);
        off += (bytes + 255) & ~(size_t)255;
        return p;
    };
    _Float16* WinT  = (_Float16*)carve((size_t)256 * 544 * 2);
    _Float16* WqT   = (_Float16*)carve((size_t)NLAYER * 256 * 256 * 2);
    _Float16* WskvT = (_Float16*)carve((size_t)NLAYER * 512 * 256 * 2);
    _Float16* WlkvT = (_Float16*)carve((size_t)NLAYER * 512 * 256 * 2);
    _Float16* WoT   = (_Float16*)carve((size_t)NLAYER * 256 * 256 * 2);
    _Float16* W1T   = (_Float16*)carve((size_t)NLAYER * 512 * 256 * 2);
    _Float16* W2T   = (_Float16*)carve((size_t)NLAYER * 256 * 512 * 2);
    _Float16* WoutT = (_Float16*)carve((size_t)V * 256 * 2);
    _Float16* cin   = (_Float16*)carve((size_t)NTOK * 544 * 2);
    float*    x     = (float*)   carve((size_t)NTOK * 256 * 4);
    _Float16* xh    = (_Float16*)carve((size_t)NTOK * 256 * 2);
    _Float16* xn    = (_Float16*)carve((size_t)NTOK * 256 * 2);
    _Float16* qh    = (_Float16*)carve((size_t)NTOK * 256 * 2);
    _Float16* skvh  = (_Float16*)carve((size_t)NTOK * 512 * 2);
    _Float16* lkvh  = (_Float16*)carve((size_t)NTOK * 512 * 2);
    _Float16* vst   = (_Float16*)carve((size_t)NTOK * 256 * 2);
    _Float16* vlt   = (_Float16*)carve((size_t)NTOK * 256 * 2);
    _Float16* aoh   = (_Float16*)carve((size_t)NTOK * 256 * 2);
    _Float16* ffh   = (_Float16*)carve((size_t)NTOK * 512 * 2);
    _Float16* xlh   = (_Float16*)carve((size_t)BATCH * 256 * 2);

    auto tr = [&](const float* W, _Float16* Wt, int K, int N) {
        k_transpose_tiled<<<dim3(K / 32, N / 32), 256, 0, stream>>>(W, Wt, K, N);
    };
    tr(Win, WinT, 544, 256);
    for (int l = 0; l < NLAYER; ++l) {
        tr(Wq   + (size_t)l * 256 * 256, WqT   + (size_t)l * 256 * 256, 256, 256);
        tr(Wskv + (size_t)l * 256 * 512, WskvT + (size_t)l * 512 * 256, 256, 512);
        tr(Wlkv + (size_t)l * 256 * 512, WlkvT + (size_t)l * 512 * 256, 256, 512);
        tr(Wo   + (size_t)l * 256 * 256, WoT   + (size_t)l * 256 * 256, 256, 256);
        tr(W1   + (size_t)l * 256 * 512, W1T   + (size_t)l * 512 * 256, 256, 512);
        tr(W2   + (size_t)l * 512 * 256, W2T   + (size_t)l * 256 * 512, 512, 256);
    }
    tr(Wout, WoutT, 256, V);

    k_embed<<<NTOK, 256, 0, stream>>>(loc, start_mn, weekday, user, duration,
                                      fine_E, coarse_E, hour_E, wd_E, Wtc, bt,
                                      lnt_g, lnt_b, user_E, Wd, bd, cin);

    auto gemm = [&](const _Float16* A, const _Float16* Wt, const float* bias,
                    const float* R, float* C, _Float16* Ch,
                    int M, int N, int K, int flags) {
        dim3 grid(M / 16, (N / 16 + 7) / 8);
        k_wmma_gemm<<<grid, 256, 0, stream>>>(A, Wt, bias, R, C, Ch, M, N, K, flags);
    };

    // input projection: x = cin @ Win + bin
    gemm(cin, WinT, bin, nullptr, x, nullptr, NTOK, 256, 544, 1);

    for (int l = 0; l < NLAYER; ++l) {
        k_ln<<<NTOK, 256, 0, stream>>>(x, ln_g + l * 256, ln_b + l * 256, xn, 0);
        gemm(xn, WqT   + (size_t)l * 256 * 256, nullptr, nullptr, nullptr, qh,   NTOK, 256, 256, 0);
        gemm(xn, WskvT + (size_t)l * 512 * 256, nullptr, nullptr, nullptr, skvh, NTOK, 512, 256, 0);
        gemm(xn, WlkvT + (size_t)l * 512 * 256, nullptr, nullptr, nullptr, lkvh, NTOK, 512, 256, 0);
        k_transpose_v<<<NTOK, 256, 0, stream>>>(skvh, vst);
        k_transpose_v<<<NTOK, 256, 0, stream>>>(lkvh, vlt);
        k_attn_wmma<<<dim3(BATCH * NHEAD, LSEQ / 128), 256, 0, stream>>>(
            qh, skvh, lkvh, vst, vlt, mix, decay, l, aoh);
        // x = x + ao @ Wo   (also f16 copy for FFN input)
        gemm(aoh, WoT + (size_t)l * 256 * 256, nullptr, x, x, xh, NTOK, 256, 256, 4);
        // h = gelu(x @ W1 + b1)
        gemm(xh, W1T + (size_t)l * 512 * 256, b1 + (size_t)l * 512, nullptr, nullptr, ffh,
             NTOK, 512, 256, 1 | 2);
        // x = x + h @ W2 + b2
        gemm(ffh, W2T + (size_t)l * 256 * 512, b2 + (size_t)l * 256, x, x, nullptr,
             NTOK, 256, 512, 1 | 4);
    }

    // head: logits = LN(x[:, -1, :]) @ Wout + bout  (single pass over WoutT)
    k_ln<<<BATCH, 256, 0, stream>>>(x, out_g, out_b, xlh, 1);
    k_head_gemm<<<(V / 16 + 7) / 8, 256, 0, stream>>>(xlh, WoutT, bout, logits, V, 256);
}
